// Module_10926396801093
// MI455X (gfx1250) — matrix operand order repaired and checked by reading, not compiled, not run on hardware
//
#include <hip/hip_runtime.h>
#include <hip/hip_bf16.h>

typedef __bf16 bf16;
typedef __attribute__((ext_vector_type(16))) __bf16 v16bf;
typedef __attribute__((ext_vector_type(8)))  __bf16 v8bf;
typedef __attribute__((ext_vector_type(4)))  __bf16 v4bf;
typedef __attribute__((ext_vector_type(8)))  float  v8f;
typedef __attribute__((ext_vector_type(4)))  float  v4f;

#define DD 128
#define PA 136   // bf16 pitch, activation tiles (272B rows: 4-bank lane stride, conflict-free b128)
#define PW 136   // bf16 pitch, K=128 weight tiles
#define PG 72    // bf16 pitch, K=64 tiles (g / W_g)
#define PH 132   // f32 pitch, residual H tile
#define WSLOT 16384  // bf16 elements per weight slot in workspace

__device__ __forceinline__ float softplus_f(float x) {
    return fmaxf(x, 0.f) + __logf(1.f + __expf(-fabsf(x)));
}

// ---- WMMA fragment loads from LDS (bf16) -------------------------------
// A 16x32 layout: lane<16 holds M=lane, K=kk..kk+7 in v0..3, K=kk+16..kk+23 in v4..7;
// lane>=16 holds K=kk+8.. / kk+24..
__device__ __forceinline__ v16bf load_a_frag(const bf16* A, int pitch, int m0, int kk, int lane) {
    int lr = lane & 15, hi = lane >> 4;
    const bf16* p = A + (m0 + lr) * pitch + kk + hi * 8;
    v8bf lo = *(const v8bf*)p;
    v8bf hh = *(const v8bf*)(p + 16);
    return __builtin_shufflevector(lo, hh, 0,1,2,3,4,5,6,7,8,9,10,11,12,13,14,15);
}
// B 32x16 layout: lane<16 holds N=lane, K=kk..kk+7 and kk+16..kk+23; lane>=16 holds K=kk+8.. / kk+24..
// W stored transposed WT[n][k] so each lane reads two 16-byte runs of k.
__device__ __forceinline__ v16bf load_b_frag(const bf16* W, int pitch, int n0, int kk, int lane) {
    int lr = lane & 15, hi = lane >> 4;
    const bf16* p = W + (n0 + lr) * pitch + kk + hi * 8;
    v8bf lo = *(const v8bf*)p;
    v8bf hh = *(const v8bf*)(p + 16);
    return __builtin_shufflevector(lo, hh, 0,1,2,3,4,5,6,7,8,9,10,11,12,13,14,15);
}

// One wave computes rows [m0, m0+16) x cols [n0, n0+16*NT), K=KLEN.
// Software-pipelined: next B (and next K-step's A) fragments are fetched from
// LDS before the current WMMA issues, so dscnt waits overlap matrix ops.
template <int NT, int KLEN>
__device__ __forceinline__ void wave_gemm(const bf16* A, int pa, const bf16* W, int pw,
                                          int m0, int n0, int lane, v8f acc[NT]) {
    v16bf a = load_a_frag(A, pa, m0, 0, lane);
    v16bf b = load_b_frag(W, pw, n0, 0, lane);
#pragma unroll
    for (int kk = 0; kk < KLEN; kk += 32) {
        v16bf a_next = a;
        if (kk + 32 < KLEN) a_next = load_a_frag(A, pa, m0, kk + 32, lane);
#pragma unroll
        for (int n = 0; n < NT; ++n) {
            v16bf b_next = b;
            if (n + 1 < NT)
                b_next = load_b_frag(W, pw, n0 + (n + 1) * 16, kk, lane);
            else if (kk + 32 < KLEN)
                b_next = load_b_frag(W, pw, n0, kk + 32, lane);
            acc[n] = __builtin_amdgcn_wmma_f32_16x16x32_bf16(
                false, a, false, b, (short)0, acc[n], false, false);
            b = b_next;
        }
        a = a_next;
    }
}

// Cooperative copy of one transposed bf16 weight matrix (128 x kdim) ws -> LDS.
__device__ __forceinline__ void stage_weights(bf16* dst, int pitch, const bf16* src,
                                              int kdim, int tid) {
    int chunks = kdim >> 3;            // 16B chunks per row
    int total = DD * chunks;
    for (int i = tid; i < total; i += 256) {
        int r = i / chunks, c = (i - r * chunks) << 3;
        *(v8bf*)(dst + r * pitch + c) = *(const v8bf*)(src + r * kdim + c);
    }
}

// ---- Kernel 0: convert all weights to transposed bf16 in workspace -----
__global__ void k_prep_weights(bf16* wb, const float* Wg, const float* Wj, const float* Wi,
                               const float* Wf, const float* Wout, const float* Wri,
                               const float* Wra, const float* Wro) {
    int s = blockIdx.x;
    const float* src;
    int K = DD;
    if (s == 0)      { src = Wg; K = 64; }
    else if (s == 1) src = Wj;
    else if (s == 2) src = Wi;
    else if (s == 3) src = Wf;
    else if (s == 4) src = Wout;
    else if (s < 9)  src = Wri + (size_t)(s - 5) * DD * DD;
    else if (s < 13) src = Wra + (size_t)(s - 9) * DD * DD;
    else             src = Wro + (size_t)(s - 13) * DD * DD;
    bf16* dst = wb + (size_t)s * WSLOT;      // WT[n][k], pitch K
    int total = K * DD;
    for (int i = threadIdx.x; i < total; i += blockDim.x) {
        int k = i / DD, n = i - k * DD;
        dst[n * K + k] = (bf16)src[(size_t)k * DD + n];
    }
}

// ---- Kernel 1: xp = u*x ; v = act(act(x)@W_i + b_i)  -------------------
__global__ __launch_bounds__(256) void k_node_pre(const float* __restrict__ x,
        const float* __restrict__ u, const bf16* __restrict__ wb,
        const float* __restrict__ b_i, float* __restrict__ xp, float* __restrict__ v, int N) {
    extern __shared__ char smem[];
    bf16* sA = (bf16*)smem;                       // 128*PA
    bf16* sW = (bf16*)(smem + 128 * PA * 2);      // 128*PW
    int tid = threadIdx.x, lane = tid & 31, wave = tid >> 5, m0 = wave * 16;
    int blk = blockIdx.x;
    {   // load 64 cols of one row per thread: act(x) -> sA (bf16), u*x -> xp
        int r = tid >> 1, hf = tid & 1;
        int grow = blk * 128 + r;
        int grc = grow < N ? grow : N - 1;
        const float* xr = x + (size_t)grc * DD + hf * 64;
        const float* ur = u + hf * 64;
        float* xpr = xp + (size_t)grc * DD + hf * 64;
        bf16* ar = sA + r * PA + hf * 64;
#pragma unroll
        for (int c = 0; c < 64; c += 4) {
            v4f xv = *(const v4f*)(xr + c);
            v4f uv = *(const v4f*)(ur + c);
            if (grow < N) *(v4f*)(xpr + c) = xv * uv;
            v4bf av;
#pragma unroll
            for (int k = 0; k < 4; ++k) av[k] = (bf16)softplus_f(xv[k]);
            *(v4bf*)(ar + c) = av;
        }
    }
    stage_weights(sW, PW, wb + 2 * WSLOT, DD, tid);   // W_i
    __syncthreads();
    v8f acc[8] = {};
    wave_gemm<8, DD>(sA, PA, sW, PW, m0, 0, lane, acc);
    int lr = lane & 15, hi = lane >> 4;
#pragma unroll
    for (int n = 0; n < 8; ++n) {
        int col = n * 16 + lr;
        float bv = b_i[col];
#pragma unroll
        for (int i = 0; i < 8; ++i) {
            int grow = blk * 128 + m0 + i + hi * 8;
            if (grow < N) v[(size_t)grow * DD + col] = softplus_f(acc[n][i] + bv);
        }
    }
}

// ---- Kernel 2: edges -> v += act(act(x_j)@W_j+b_j) * (g@W_g+b_g) -------
__global__ __launch_bounds__(256) void k_edge(const float* __restrict__ x,
        const float* __restrict__ g, const int* __restrict__ idx_i,
        const int* __restrict__ idx_j, const bf16* __restrict__ wb,
        const float* __restrict__ b_j, const float* __restrict__ b_g,
        float* __restrict__ v, int E) {
    extern __shared__ char smem[];
    bf16* sA  = (bf16*)smem;                                    // 128*PA  act(x_j)
    bf16* sWj = (bf16*)(smem + 128 * PA * 2);                   // 128*PW
    bf16* sG  = (bf16*)(smem + 128 * (PA + PW) * 2);            // 128*PG  g tile
    bf16* sWg = (bf16*)(smem + 128 * (PA + PW + PG) * 2);       // 128*PG
    int tid = threadIdx.x, lane = tid & 31, wave = tid >> 5, m0 = wave * 16;
    int e0 = blockIdx.x * 128;
    {
        int r = tid >> 1, hf = tid & 1;
        int e = e0 + r; if (e >= E) e = E - 1;
        int j = idx_j[e];
        const float* xr = x + (size_t)j * DD + hf * 64;
        bf16* ar = sA + r * PA + hf * 64;
#pragma unroll
        for (int c = 0; c < 64; c += 4) {
            v4f xv = *(const v4f*)(xr + c);
            v4bf av;
#pragma unroll
            for (int k = 0; k < 4; ++k) av[k] = (bf16)softplus_f(xv[k]);
            *(v4bf*)(ar + c) = av;
        }
        const float* gr = g + (size_t)e * 64 + hf * 32;
        bf16* gl = sG + r * PG + hf * 32;
#pragma unroll
        for (int c = 0; c < 32; c += 4) {
            v4f gv = *(const v4f*)(gr + c);
            v4bf av;
#pragma unroll
            for (int k = 0; k < 4; ++k) av[k] = (bf16)gv[k];
            *(v4bf*)(gl + c) = av;
        }
    }
    stage_weights(sWj, PW, wb + 1 * WSLOT, DD, tid);  // W_j
    stage_weights(sWg, PG, wb + 0 * WSLOT, 64, tid);  // W_g
    __syncthreads();
    int lr = lane & 15, hi = lane >> 4;
#pragma unroll 1
    for (int nh = 0; nh < 2; ++nh) {              // split N=128 into halves -> fewer live accs
        v8f accP[4] = {};
        v8f accQ[4] = {};
        wave_gemm<4, DD>(sA, PA, sWj, PW, m0, nh * 64, lane, accP);
        wave_gemm<4, 64>(sG, PG, sWg, PG, m0, nh * 64, lane, accQ);
#pragma unroll
        for (int i = 0; i < 8; ++i) {
            int erow = e0 + m0 + i + hi * 8;
            if (erow < E) {
                int seg = idx_i[erow];
                float* vrow = v + (size_t)seg * DD;
#pragma unroll
                for (int n = 0; n < 4; ++n) {
                    int col = nh * 64 + n * 16 + lr;
                    float pv = softplus_f(accP[n][i] + b_j[col]);
                    float qv = accQ[n][i] + b_g[col];
                    atomicAdd(vrow + col, pv * qv);   // fp32 segment-sum in L2
                }
            }
        }
    }
}

// ---- Fused node chain helpers ------------------------------------------
// Invariant: sA == bf16(softplus(sH)) for this wave's rows on entry/exit.
__device__ __forceinline__ void res_block(float* sH, bf16* sA, bf16* sW,
        const bf16* wb, int slot0, const float* b0, const float* b1,
        int tid, int lane, int m0) {
    int lr = lane & 15, hi = lane >> 4;
    __syncthreads();
    stage_weights(sW, PW, wb + (size_t)slot0 * WSLOT, DD, tid);
    __syncthreads();
    v8f acc[8] = {};
    wave_gemm<8, DD>(sA, PA, sW, PW, m0, 0, lane, acc);
#pragma unroll
    for (int n = 0; n < 8; ++n) {
        int col = n * 16 + lr;
        float bv = b0[col];
#pragma unroll
        for (int i = 0; i < 8; ++i) {
            int row = m0 + i + hi * 8;
            sA[row * PA + col] = (bf16)softplus_f(acc[n][i] + bv);  // A := act(t1), own rows
        }
    }
    __syncthreads();
    stage_weights(sW, PW, wb + (size_t)(slot0 + 1) * WSLOT, DD, tid);
    __syncthreads();
    v8f acc2[8] = {};
    wave_gemm<8, DD>(sA, PA, sW, PW, m0, 0, lane, acc2);
#pragma unroll
    for (int n = 0; n < 8; ++n) {
        int col = n * 16 + lr;
        float bv = b1[col];
#pragma unroll
        for (int i = 0; i < 8; ++i) {
            int row = m0 + i + hi * 8;
            float hn = sH[row * PH + col] + acc2[n][i] + bv;        // h += t2
            sH[row * PH + col] = hn;
            sA[row * PA + col] = (bf16)softplus_f(hn);              // restore invariant
        }
    }
}

// ---- Kernel 3: residual_int -> W_f(+xp) -> residual_atom -> h ->
//                residual_out -> W_out -> o ------------------------------
__global__ __launch_bounds__(256) void k_node_main(const bf16* __restrict__ wb,
        const float* __restrict__ v, const float* __restrict__ xp,
        const float* __restrict__ bri, const float* __restrict__ bra,
        const float* __restrict__ bro, const float* __restrict__ b_f,
        const float* __restrict__ b_out, float* __restrict__ out_o,
        float* __restrict__ out_h, int N) {
    extern __shared__ char smem[];
    float* sH = (float*)smem;                                   // 128*PH f32
    bf16* sA = (bf16*)(smem + 128 * PH * 4);                    // 128*PA
    bf16* sW = (bf16*)(smem + 128 * PH * 4 + 128 * PA * 2);     // 128*PW
    int tid = threadIdx.x, lane = tid & 31, wave = tid >> 5, m0 = wave * 16;
    int blk = blockIdx.x;
    int lr = lane & 15, hi = lane >> 4;
    {   // load v tile: H=f32, A=act(H) bf16
        int r = tid >> 1, hf = tid & 1;
        int grow = blk * 128 + r;
        int grc = grow < N ? grow : N - 1;
        const float* vr = v + (size_t)grc * DD + hf * 64;
        float* hr = sH + r * PH + hf * 64;
        bf16* ar = sA + r * PA + hf * 64;
#pragma unroll
        for (int c = 0; c < 64; c += 4) {
            v4f val = *(const v4f*)(vr + c);
            *(v4f*)(hr + c) = val;
            v4bf av;
#pragma unroll
            for (int k = 0; k < 4; ++k) av[k] = (bf16)softplus_f(val[k]);
            *(v4bf*)(ar + c) = av;
        }
    }
    // interaction residual stack (2 blocks: slots 5..8)
    res_block(sH, sA, sW, wb, 5, bri + 0 * DD, bri + 1 * DD, tid, lane, m0);
    res_block(sH, sA, sW, wb, 7, bri + 2 * DD, bri + 3 * DD, tid, lane, m0);
    // h = xp + act(v)@W_f + b_f   (A already holds act(H))
    __syncthreads();
    stage_weights(sW, PW, wb + 3 * WSLOT, DD, tid);  // W_f
    __syncthreads();
    {
        v8f acc[8] = {};
        wave_gemm<8, DD>(sA, PA, sW, PW, m0, 0, lane, acc);
#pragma unroll
        for (int n = 0; n < 8; ++n) {
            int col = n * 16 + lr;
            float bv = b_f[col];
#pragma unroll
            for (int i = 0; i < 8; ++i) {
                int row = m0 + i + hi * 8;
                int grow = blk * 128 + row;
                int grc = grow < N ? grow : N - 1;
                float hn = xp[(size_t)grc * DD + col] + acc[n][i] + bv;
                sH[row * PH + col] = hn;
                sA[row * PA + col] = (bf16)softplus_f(hn);
            }
        }
    }
    // atomic residual stack (2 blocks: slots 9..12)
    res_block(sH, sA, sW, wb, 9,  bra + 0 * DD, bra + 1 * DD, tid, lane, m0);
    res_block(sH, sA, sW, wb, 11, bra + 2 * DD, bra + 3 * DD, tid, lane, m0);
    // store h (second output)
    {
        int r = tid >> 1, hf = tid & 1;
        int grow = blk * 128 + r;
        if (grow < N) {
            const float* hr = sH + r * PH + hf * 64;
            float* dst = out_h + (size_t)grow * DD + hf * 64;
#pragma unroll
            for (int c = 0; c < 64; c += 4) *(v4f*)(dst + c) = *(const v4f*)(hr + c);
        }
    }
    // output residual stack (1 block: slots 13..14)
    res_block(sH, sA, sW, wb, 13, bro + 0 * DD, bro + 1 * DD, tid, lane, m0);
    // o = act(o)@W_out + b_out  (A holds act(H))
    __syncthreads();
    stage_weights(sW, PW, wb + 4 * WSLOT, DD, tid);  // W_out
    __syncthreads();
    {
        v8f acc[8] = {};
        wave_gemm<8, DD>(sA, PA, sW, PW, m0, 0, lane, acc);
#pragma unroll
        for (int n = 0; n < 8; ++n) {
            int col = n * 16 + lr;
            float bv = b_out[col];
#pragma unroll
            for (int i = 0; i < 8; ++i) {
                int grow = blk * 128 + m0 + i + hi * 8;
                if (grow < N) out_o[(size_t)grow * DD + col] = acc[n][i] + bv;
            }
        }
    }
}

extern "C" void kernel_launch(void* const* d_in, const int* in_sizes, int n_in,
                              void* d_out, int out_size, void* d_ws, size_t ws_size,
                              hipStream_t stream) {
    (void)n_in; (void)out_size; (void)ws_size;
    const float* x     = (const float*)d_in[0];
    const float* g     = (const float*)d_in[1];
    const int*   idx_i = (const int*)d_in[2];
    const int*   idx_j = (const int*)d_in[3];
    const float* u     = (const float*)d_in[5];
    const float* W_g   = (const float*)d_in[6];  const float* b_g  = (const float*)d_in[7];
    const float* W_j   = (const float*)d_in[8];  const float* b_j  = (const float*)d_in[9];
    const float* W_i   = (const float*)d_in[10]; const float* b_i  = (const float*)d_in[11];
    const float* W_f   = (const float*)d_in[12]; const float* b_f  = (const float*)d_in[13];
    const float* Wri   = (const float*)d_in[14]; const float* bri  = (const float*)d_in[15];
    const float* Wra   = (const float*)d_in[16]; const float* bra  = (const float*)d_in[17];
    const float* Wro   = (const float*)d_in[18]; const float* bro  = (const float*)d_in[19];
    const float* W_out = (const float*)d_in[20]; const float* b_out = (const float*)d_in[21];
    int N = in_sizes[0] / DD;
    int E = in_sizes[2];
    float* out_o = (float*)d_out;
    float* out_h = (float*)d_out + (size_t)N * DD;
    // workspace layout: [15 bf16 weight slots][xp f32 N*D][v f32 N*D]
    bf16* wb = (bf16*)d_ws;
    float* xp = (float*)((char*)d_ws + (size_t)15 * WSLOT * 2);
    float* vv = xp + (size_t)N * DD;
    int nblk = (N + 127) / 128;
    int eblk = (E + 127) / 128;
    size_t lds_pre  = (size_t)128 * (PA + PW) * 2;
    size_t lds_edge = (size_t)128 * (PA + PW + 2 * PG) * 2;
    size_t lds_main = (size_t)128 * PH * 4 + (size_t)128 * (PA + PW) * 2;
    k_prep_weights<<<15, 256, 0, stream>>>(wb, W_g, W_j, W_i, W_f, W_out, Wri, Wra, Wro);
    k_node_pre<<<nblk, 256, lds_pre, stream>>>(x, u, wb, b_i, xp, vv, N);
    k_edge<<<eblk, 256, lds_edge, stream>>>(x, g, idx_i, idx_j, wb, b_j, b_g, vv, E);
    k_node_main<<<nblk, 256, lds_main, stream>>>(wb, vv, xp, bri, bra, bro, b_f, b_out,
                                                 out_o, out_h, N);
}
